// LSTM_55456617726021
// MI455X (gfx1250) — matrix operand order repaired and checked by reading, not compiled, not run on hardware
//
#include <hip/hip_runtime.h>
#include <hip/hip_bf16.h>
#include <stdint.h>
#include <stddef.h>

// ---- problem sizes --------------------------------------------------------
#define B_  64
#define T_  512
#define H_  256
#define L_  2
#define G4  1024      // 4*H
#define V_  32000

typedef __bf16 bf16;
typedef __attribute__((ext_vector_type(16))) __bf16 v16bf;
typedef __attribute__((ext_vector_type(8)))  __bf16 v8bf;
typedef __attribute__((ext_vector_type(8)))  float  v8f;
typedef __attribute__((ext_vector_type(4)))  float  v4f;
typedef __attribute__((ext_vector_type(4)))  unsigned int v4u;

// D = A(16x32 bf16) * B(32x16 bf16) + C(16x16 f32)
__device__ __forceinline__ v8f wmma_bf16(v16bf a, v16bf b, v8f c) {
  return __builtin_amdgcn_wmma_f32_16x16x32_bf16(
      /*neg_a=*/false, a, /*neg_b=*/false, b,
      /*c_mod=*/(short)0, c, /*reuse_a=*/false, /*reuse_b=*/false);
}

// A fragment: lane m = lane&15, k-half kh = lane>>4.
__device__ __forceinline__ v16bf load_afrag(const bf16* row, int k0, int kh) {
  const bf16* p = row + k0 + kh * 16;
  v8bf a0 = *(const v8bf*)p;
  v8bf a1 = *(const v8bf*)(p + 8);
  return __builtin_shufflevector(a0, a1, 0,1,2,3,4,5,6,7,8,9,10,11,12,13,14,15);
}

// ---- CDNA5 async LDS<->global copies (ASYNCcnt tracked) -------------------
// Low 32 bits of a generic shared pointer are the LDS byte address (ISA 10.2).
__device__ __forceinline__ void async_ld_b128(const void* g, void* lds) {
  unsigned loff = (unsigned)(size_t)lds;
  asm volatile("global_load_async_to_lds_b128 %0, %1, off"
               :: "v"(loff), "v"(g) : "memory");
}
__device__ __forceinline__ void async_ld_b128_dev(const void* g, void* lds) {
  unsigned loff = (unsigned)(size_t)lds;
  asm volatile("global_load_async_to_lds_b128 %0, %1, off scope:SCOPE_DEV"
               :: "v"(loff), "v"(g) : "memory");
}
__device__ __forceinline__ void async_st_b128_dev(void* g, const void* lds) {
  unsigned loff = (unsigned)(size_t)lds;
  asm volatile("global_store_async_from_lds_b128 %0, %1, off scope:SCOPE_DEV"
               :: "v"(g), "v"(loff) : "memory");
}
__device__ __forceinline__ void wait_async0() {
  asm volatile("s_wait_asynccnt 0x0" ::: "memory");
}

// ---- prep: bf16 conversions, combined bias, h0 staging, barrier reset -----
__global__ void lstm_prep(const float* __restrict__ emb, const float* __restrict__ Wih,
                          const float* __restrict__ Whh, const float* __restrict__ bih,
                          const float* __restrict__ bhh, const float* __restrict__ h0,
                          bf16* __restrict__ embB, bf16* __restrict__ wihB,
                          bf16* __restrict__ whhB, float* __restrict__ bsum,
                          bf16* __restrict__ hbuf, unsigned* __restrict__ bars) {
  int i = blockIdx.x * blockDim.x + threadIdx.x;
  if (i < V_ * H_)    embB[i] = (bf16)emb[i];
  if (i < L_ * G4 * H_) { wihB[i] = (bf16)Wih[i]; whhB[i] = (bf16)Whh[i]; }
  if (i < L_ * G4)    bsum[i] = bih[i] + bhh[i];
  if (i < L_ * B_ * H_) {                         // hbuf[l][slot0] = h0[l]
    int l = i / (B_ * H_), j = i % (B_ * H_);
    hbuf[(l * 2 + 0) * (B_ * H_) + j] = (bf16)h0[i];
  }
  if (i < 128) bars[i] = 0u;                      // per-layer barrier counters
}

// ---- pre-GEMM: pre[m, n] = act[m,:] . Wih[n,:] + bsum[n] ------------------
__global__ void __launch_bounds__(256) lstm_pregemm(
    const int* __restrict__ x, const bf16* __restrict__ embB,
    const bf16* __restrict__ y0, const bf16* __restrict__ wihB,
    const float* __restrict__ bsum, float* __restrict__ prebuf, int layer) {
  const int lane = threadIdx.x & 31;
  const int wv   = threadIdx.x >> 5;
  const int wave_id = blockIdx.x * 8 + wv;        // 32768 waves total
  const int mt = wave_id >> 4;                    // 0..2047  (M tile)
  const int n0 = (wave_id & 15) * 64;             // N strip base
  const int kh = lane >> 4;
  const int m0 = mt * 16;
  const int m  = m0 + (lane & 15);
  const int t  = m >> 6, b = m & 63;

  const bf16* arow = (layer == 0)
      ? (embB + (size_t)x[b * T_ + t] * H_)       // embedding gather row
      : (y0   + (size_t)m * H_);                  // previous layer output
  const bf16* wihL = wihB + (size_t)layer * G4 * H_;

  v8f acc[4];
  const bf16* brow[4];
  for (int nt = 0; nt < 4; ++nt) {
    int n = n0 + nt * 16 + (lane & 15);
    brow[nt] = wihL + (size_t)n * H_;
    float bv = bsum[layer * G4 + n];
    acc[nt] = (v8f){bv, bv, bv, bv, bv, bv, bv, bv};
  }
  for (int kb = 0; kb < 8; ++kb) {
    int k0 = kb * 32;
    v16bf a = load_afrag(arow, k0, kh);
    for (int nt = 0; nt < 4; ++nt) {
      v16bf bfrag = *(const v16bf*)(brow[nt] + k0 + kh * 16);
      acc[nt] = wmma_bf16(a, bfrag, acc[nt]);
    }
  }
  // store: D[m=batch, n=gate-row] -> prebuf[t][wg][r=gate*16+u][b], b contiguous
  const int tq = m0 >> 6;
  const int bb = (m0 & 63) + kh * 8;
  for (int nt = 0; nt < 4; ++nt) {
    int n = n0 + nt * 16 + (lane & 15);
    int gate = n >> 8, wg = (n >> 4) & 15, u = n & 15;
    int r = gate * 16 + u;
    float* p = prebuf + ((((size_t)tq * 16 + wg) * 64 + r) * 64 + bb);
    *(v4f*)p       = __builtin_shufflevector(acc[nt], acc[nt], 0, 1, 2, 3);
    *(v4f*)(p + 4) = __builtin_shufflevector(acc[nt], acc[nt], 4, 5, 6, 7);
  }
}

// ---- persistent recurrent scan: 16 WGs, one per 16 hidden units -----------
__global__ void __launch_bounds__(256) lstm_scan(
    const float* __restrict__ prebuf, const bf16* __restrict__ whhB,
    const float* __restrict__ c0, bf16* __restrict__ hbuf,
    bf16* __restrict__ y0, float* __restrict__ out,
    float* __restrict__ hT, float* __restrict__ cT,
    unsigned* __restrict__ bar, int layer) {
  __shared__ alignas(32) bf16  s_whh[64 * 256];   // 32 KB: this WG's Whh rows
  __shared__ alignas(32) bf16  s_h[64 * 256];     // 32 KB: full h state (bf16)
  __shared__ alignas(32) float s_g[64 * 64];      // 16 KB: gate pre-acts [r][b]
  __shared__ alignas(32) bf16  s_hs[64 * 16];     //  2 KB: h slice bf16 [b][u]
  __shared__ alignas(32) float s_ho[64 * 16];     //  4 KB: h slice f32  [b][u]

  const int wg = blockIdx.x;
  const int tid = threadIdx.x;
  const int lane = tid & 31, wv = tid >> 5, kh = lane >> 4;

  // stage Whh slice (reordered: r = gate*16+u <- row gate*256+wg*16+u) + h0,
  // through the async LDS engine (no VGPR round-trip).
  const bf16* whhL = whhB + (size_t)layer * G4 * H_;
  for (int i = tid; i < 2048; i += 256) {         // 2048 x 16B = 32 KB
    int r = i >> 5, off = (i & 31) * 8;
    int G = (r >> 4) * 256 + wg * 16 + (r & 15);
    async_ld_b128(&whhL[(size_t)G * 256 + off], &s_whh[r * 256 + off]);
  }
  bf16* hb = hbuf + (size_t)layer * 2 * (B_ * H_);
  for (int i = tid; i < 2048; i += 256)           // slot 0 holds h0
    async_ld_b128(&hb[i * 8], &s_h[i * 8]);

  float c_reg[4];
  for (int q = 0; q < 4; ++q) {                   // 4 (b,u) cells per thread
    int idx = tid + 256 * q, b = idx & 63, u = idx >> 6;
    c_reg[q] = c0[(size_t)layer * B_ * H_ + b * H_ + wg * 16 + u];
  }
  wait_async0();
  __syncthreads();

  for (int t = 0; t < T_; ++t) {
    const float* pre_t = prebuf + ((size_t)t * 16 + wg) * 4096;   // 16 KB block
    // g = pre_t + h @ Whh_slice^T : 16 tiles, 2 per wave, K=256
    for (int i = 0; i < 2; ++i) {
      int tile = wv * 2 + i, mt = tile >> 2, nt = tile & 3;
      int rc = nt * 16 + (lane & 15);             // local gate-row (D column)
      int b0 = mt * 16 + kh * 8;                  // batch base (D rows)
      const float* pc = pre_t + rc * 64 + b0;
      v4f cl = *(const v4f*)pc, chv = *(const v4f*)(pc + 4);
      v8f acc = __builtin_shufflevector(cl, chv, 0, 1, 2, 3, 4, 5, 6, 7);
      const bf16* arow = &s_h[(mt * 16 + (lane & 15)) * 256];
      const bf16* brow = &s_whh[rc * 256];
      for (int kb = 0; kb < 8; ++kb) {
        int k0 = kb * 32;
        v16bf a = load_afrag(arow, k0, kh);
        v16bf bfrag = *(const v16bf*)(brow + k0 + kh * 16);
        acc = wmma_bf16(a, bfrag, acc);
      }
      float* pg = s_g + rc * 64 + b0;
      *(v4f*)pg       = __builtin_shufflevector(acc, acc, 0, 1, 2, 3);
      *(v4f*)(pg + 4) = __builtin_shufflevector(acc, acc, 4, 5, 6, 7);
    }
    __syncthreads();

    if (t + 1 < T_) {                             // prefetch next pre block
      const char* nx = (const char*)(prebuf + ((size_t)(t + 1) * 16 + wg) * 4096);
      __builtin_prefetch(nx + tid * 64, 0, 1);
    }

    int slot_w = (t + 1) & 1;                     // double-buffered h broadcast
    bf16* hw = hb + slot_w * (B_ * H_);
    for (int q = 0; q < 4; ++q) {
      int idx = tid + 256 * q, b = idx & 63, u = idx >> 6;
      float gi = s_g[u * 64 + b];                 // PyTorch order: i,f,g,o
      float gf = s_g[(16 + u) * 64 + b];
      float gc = s_g[(32 + u) * 64 + b];
      float go = s_g[(48 + u) * 64 + b];
      float iv = 1.f / (1.f + __expf(-gi));
      float fv = 1.f / (1.f + __expf(-gf));
      float ov = 1.f / (1.f + __expf(-go));
      float cv = tanhf(gc);
      c_reg[q] = fv * c_reg[q] + iv * cv;
      float h  = ov * tanhf(c_reg[q]);
      s_hs[b * 16 + u] = (bf16)h;                 // stage slice in LDS
      s_ho[b * 16 + u] = h;
      if (t == T_ - 1) {
        int col = wg * 16 + u;
        hT[b * H_ + col] = h; cT[b * H_ + col] = c_reg[q];
      }
    }
    __syncthreads();                              // slice staged

    // broadcast h slice (and y outputs) with coalesced async stores, DEV scope
    if (tid < 128) {
      int b = tid >> 1, hh = tid & 1;
      const bf16* src = &s_hs[b * 16 + hh * 8];
      async_st_b128_dev(hw + b * H_ + wg * 16 + hh * 8, src);
      if (layer == 0)
        async_st_b128_dev(y0 + ((size_t)t * 64 + b) * H_ + wg * 16 + hh * 8, src);
    }
    if (layer == 1) {
      int b = tid >> 2, j = tid & 3;
      async_st_b128_dev(out + ((size_t)b * T_ + t) * H_ + wg * 16 + j * 4,
                        &s_ho[b * 16 + j * 4]);
    }
    wait_async0();                                // stores complete at DEV scope
    __threadfence();
    __syncthreads();
    if (tid == 0) {                               // one device barrier per step
      __hip_atomic_fetch_add(bar, 1u, __ATOMIC_ACQ_REL, __HIP_MEMORY_SCOPE_AGENT);
      unsigned target = 16u * (unsigned)(t + 1);
      while (__hip_atomic_load(bar, __ATOMIC_ACQUIRE, __HIP_MEMORY_SCOPE_AGENT) < target)
        __builtin_amdgcn_s_sleep(1);
    }
    __syncthreads();
    // restage full h for t+1: async copy at DEV scope (coherent vs other WGPs)
    for (int i = tid; i < 2048; i += 256)
      async_ld_b128_dev(&hw[i * 8], &s_h[i * 8]);
    wait_async0();
    __syncthreads();
  }
}

// ---- host ----------------------------------------------------------------
extern "C" void kernel_launch(void* const* d_in, const int* in_sizes, int n_in,
                              void* d_out, int out_size, void* d_ws, size_t ws_size,
                              hipStream_t stream) {
  (void)in_sizes; (void)n_in; (void)out_size; (void)ws_size;
  const int*   x   = (const int*)  d_in[0];
  const float* emb = (const float*)d_in[1];
  const float* Wih = (const float*)d_in[2];
  const float* Whh = (const float*)d_in[3];
  const float* bih = (const float*)d_in[4];
  const float* bhh = (const float*)d_in[5];
  const float* h0  = (const float*)d_in[6];
  const float* c0  = (const float*)d_in[7];
  float* outp = (float*)d_out;

  char* ws = (char*)d_ws;
  bf16*     embB   = (bf16*)    (ws + 0);           // 16,384,000 B
  bf16*     wihB   = (bf16*)    (ws + 16384000);    //  1,048,576 B
  bf16*     whhB   = (bf16*)    (ws + 17432576);    //  1,048,576 B
  float*    bsum   = (float*)   (ws + 18481152);    //      8,192 B
  bf16*     hbuf   = (bf16*)    (ws + 18489344);    //    131,072 B  [L][2][B][H]
  unsigned* bars   = (unsigned*)(ws + 18620416);    //        512 B
  bf16*     y0     = (bf16*)    (ws + 18620928);    // 16,777,216 B  [T*64+b][H]
  float*    prebuf = (float*)   (ws + 35398144);    //134,217,728 B  [T][16][64][64]

  float* out_y = outp;                              // [B,T,H]
  float* out_h = outp + (size_t)B_ * T_ * H_;       // [L,B,H]
  float* out_c = out_h + (size_t)L_ * B_ * H_;      // [L,B,H]

  lstm_prep<<<(V_ * H_ + 255) / 256, 256, 0, stream>>>(
      emb, Wih, Whh, bih, bhh, h0, embB, wihB, whhB, bsum, hbuf, bars);

  for (int l = 0; l < L_; ++l) {
    lstm_pregemm<<<4096, 256, 0, stream>>>(x, embB, y0, wihB, bsum, prebuf, l);
    lstm_scan<<<16, 256, 0, stream>>>(prebuf, whhB, c0, hbuf, y0, out_y,
                                      out_h + (size_t)l * B_ * H_,
                                      out_c + (size_t)l * B_ * H_,
                                      bars + l * 64, l);
  }
}
